// PI0Pytorch_77421080478475
// MI455X (gfx1250) — matrix operand order repaired and checked by reading, not compiled, not run on hardware
//
#include <hip/hip_runtime.h>

// ---------------------------------------------------------------------------
// Problem constants (from reference)
// ---------------------------------------------------------------------------
#define BB   4
#define S1N  768
#define S2N  64
#define SN   (S1N + S2N)        // 832
#define D0N  2048
#define F0N  16384
#define D1N  1024
#define F1N  4096
#define HN   8
#define HDN  256
#define MASK_VAL (-1.0e9f)

#define LDSS 34                  // LDS row stride in halves (64B row + 4B TDM pad)

typedef __attribute__((ext_vector_type(16))) __bf16 v16bf;
typedef __attribute__((ext_vector_type(8)))  float  v8f;

union Frag { v16bf v; unsigned int u[8]; };

__device__ __forceinline__ unsigned short f2bf(float f) {
    unsigned int u = __builtin_bit_cast(unsigned int, f);
    u += 0x7FFFu + ((u >> 16) & 1u);            // round-to-nearest-even
    return (unsigned short)(u >> 16);
}
__device__ __forceinline__ float bf2f(unsigned short h) {
    unsigned int u = ((unsigned int)h) << 16;
    return __builtin_bit_cast(float, u);
}

#define HAVE_TDM __has_builtin(__builtin_amdgcn_tensor_load_to_lds)

#if HAVE_TDM
typedef unsigned int u32x4 __attribute__((ext_vector_type(4)));
typedef int          i32x8 __attribute__((ext_vector_type(8)));
typedef int          i32x4 __attribute__((ext_vector_type(4)));

// Issue one TDM 2D tile load: [128 rows x 32 halves] from row-major global
// (row stride = strideElems halves) into LDS at ldsByte, with 1-dword pad
// after each 16-dword (64B) row => LDS row stride = 34 halves.
// rowsRem/kRem clamp via tensor_dim -> hardware zero-fills OOB.
__device__ __forceinline__ void tdm_tile_load(const unsigned short* gptr,
                                              unsigned ldsByte,
                                              int rowsRem, int kRem, int strideElems)
{
    unsigned long long ga = (unsigned long long)gptr;
    u32x4 g0;
    g0.x = 1u;                                              // count=1 (valid)
    g0.y = ldsByte;                                         // lds_addr
    g0.z = (unsigned)ga;                                    // global_addr[31:0]
    g0.w = ((unsigned)(ga >> 32) & 0x01FFFFFFu) | (2u << 30); // [56:32] | type=2
    const unsigned td0 = (unsigned)kRem;                    // tensor_dim0
    const unsigned td1 = (unsigned)(rowsRem > 0 ? rowsRem : 0); // tensor_dim1
    const unsigned long long st = (unsigned long long)(unsigned)strideElems;
    i32x8 g1;
    g1[0] = (int)0x00D10000u;  // data_size=2B | pad_enable | pad_interval=16dw | pad_amount=1dw
    g1[1] = (int)((td0 & 0xFFFFu) << 16);                   // tensor_dim0[15:0]
    g1[2] = (int)((td0 >> 16) | ((td1 & 0xFFFFu) << 16));   // dim0 hi | dim1 lo
    g1[3] = (int)(((td1 >> 16) & 0xFFFFu) | (32u << 16));   // dim1 hi | tile_dim0=32
    g1[4] = 128;                                            // tile_dim1=128, tile_dim2=0
    g1[5] = (int)(unsigned)st;                              // tensor_dim0_stride lo
    g1[6] = (int)(unsigned)((st >> 32) & 0xFFFFu);          // stride hi | dim1_stride=0
    g1[7] = 0;
    i32x4 gz;  gz[0] = gz[1] = gz[2] = gz[3] = 0;           // 2D: groups 2/3 unused
    i32x8 gz8; gz8[0] = gz8[1] = gz8[2] = gz8[3] = gz8[4] = gz8[5] = gz8[6] = gz8[7] = 0;
    // clang-23 / therock-10.0 arity: (g0, g1, g2, g3, i32x8, cpol)
    __builtin_amdgcn_tensor_load_to_lds(g0, g1, gz, gz, gz8, 0);
}
#endif

// ---------------------------------------------------------------------------
// Shared WMMA inner step: one 32-deep K slice from LDS tiles (stride LDSS)
// ---------------------------------------------------------------------------
__device__ __forceinline__ void wmma_step(const unsigned short* __restrict__ as,
                                          const unsigned short* __restrict__ bs,
                                          int wm, int wn, int lm, int lh,
                                          v8f acc[2][4])
{
    Frag af[2], bf[4];
    #pragma unroll
    for (int im = 0; im < 2; ++im) {
        const int row = wm * 32 + im * 16 + lm;
        #pragma unroll
        for (int v = 0; v < 8; ++v) {
            const int kk = (v >> 2) * 16 + lh * 8 + (v & 3) * 2;   // ISA A 16x32 layout
            af[im].u[v] = *(const unsigned int*)&as[row * LDSS + kk];
        }
    }
    #pragma unroll
    for (int in_ = 0; in_ < 4; ++in_) {
        const int col = wn * 64 + in_ * 16 + lm;
        #pragma unroll
        for (int v = 0; v < 8; ++v) {
            const int kk = (v >> 2) * 16 + lh * 8 + (v & 3) * 2;   // same k order as A
            bf[in_].u[v] = *(const unsigned int*)&bs[col * LDSS + kk];
        }
    }
    #pragma unroll
    for (int im = 0; im < 2; ++im)
        #pragma unroll
        for (int in_ = 0; in_ < 4; ++in_)
            acc[im][in_] = __builtin_amdgcn_wmma_f32_16x16x32_bf16(
                false, af[im].v, false, bf[in_].v,
                (short)0, acc[im][in_], false, false);
}

// ---------------------------------------------------------------------------
// WMMA GEMM:  C[z] = alpha * A[z] (MxK, bf16, row-major) * B[z]
//   B is ALWAYS stored [N][K] (k-contiguous): element B[k][n] at n*ldb + k.
//   OUT_BF16: store C as bf16 (else fp32)
//   CT      : store C transposed (cidx = offC + col*ldc + row), bf16 only
// Batch offsets: off = (z/div)*strideOuter + (z%div)*strideInner
// Requires K % 32 == 0 (all call sites: 2048,1024,256,832,16384,4096)
// Tile staging via Tensor Data Mover (double-buffered), wave 0 drives DMA.
// ---------------------------------------------------------------------------
template<bool OUT_BF16, bool CT>
__global__ __launch_bounds__(256)
void gemm_wmma(const unsigned short* __restrict__ A,
               const unsigned short* __restrict__ Bm,
               void* __restrict__ Cout,
               const float* __restrict__ Res,
               const float* __restrict__ Gate,
               int M, int N, int K,
               int lda, int ldb, int ldc, float alpha,
               int divA, long sAb, long sAh,
               int divB, long sBb, long sBh,
               int divC, long sCb, long sCh,
               int gdiv, int gstride)
{
    __shared__ unsigned short As[2][128][LDSS];
    __shared__ unsigned short Bs[2][128][LDSS];

    const int tid  = threadIdx.x;
    const int z    = blockIdx.z;
    const long offA = (long)(z / divA) * sAb + (long)(z % divA) * sAh;
    const long offB = (long)(z / divB) * sBb + (long)(z % divB) * sBh;
    const long offC = (long)(z / divC) * sCb + (long)(z % divC) * sCh;

    const int mBase = blockIdx.y * 128;
    const int nBase = blockIdx.x * 128;

    const int wid  = tid >> 5;
    const int lane = tid & 31;
    const int wm   = wid >> 1;        // 0..3 -> 32-row strip
    const int wn   = wid & 1;         // 0..1 -> 64-col strip
    const int lm   = lane & 15;
    const int lh   = lane >> 4;

    v8f acc[2][4];
    for (int im = 0; im < 2; ++im)
        for (int in_ = 0; in_ < 4; ++in_)
            for (int j = 0; j < 8; ++j) acc[im][in_][j] = 0.0f;

    const unsigned short* aTile = A  + offA + (long)mBase * lda;
    const unsigned short* bTile = Bm + offB + (long)nBase * ldb;
    const int nT = K >> 5;

#if HAVE_TDM
    const unsigned aOff0 = (unsigned)(unsigned long long)&As[0][0][0];
    const unsigned bOff0 = (unsigned)(unsigned long long)&Bs[0][0][0];
    const unsigned bufBytes = 128u * LDSS * 2u;

    if (wid == 0) {                                    // prologue: tile 0 -> buf 0
        tdm_tile_load(aTile, aOff0, M - mBase, K, lda);
        tdm_tile_load(bTile, bOff0, N - nBase, K, ldb);
    }
    for (int kt = 0; kt < nT; ++kt) {
        const int cur = kt & 1;
        if (wid == 0) __builtin_amdgcn_s_wait_tensorcnt(0);
        __syncthreads();                               // LDS[cur] ready, LDS[cur^1] free
        if (wid == 0 && kt + 1 < nT) {
            const int k1 = (kt + 1) << 5;
            tdm_tile_load(aTile + k1, aOff0 + (cur ^ 1) * bufBytes, M - mBase, K - k1, lda);
            tdm_tile_load(bTile + k1, bOff0 + (cur ^ 1) * bufBytes, N - nBase, K - k1, ldb);
        }
        wmma_step(&As[cur][0][0], &Bs[cur][0][0], wm, wn, lm, lh, acc);
    }
#else
    // Fallback: cooperative global->LDS staging (same LDSS layout)
    for (int kt = 0; kt < nT; ++kt) {
        const int cur = kt & 1;
        const int k0  = kt << 5;
        {   // A tile: 2 threads per row, 16 halves each
            const int r  = tid >> 1;
            const int c0 = (tid & 1) * 16;
            unsigned int* dst = (unsigned int*)&As[cur][r][c0];
            if (mBase + r < M) {
                const unsigned short* ap = aTile + (long)r * lda + k0 + c0;
                #pragma unroll
                for (int i = 0; i < 8; ++i) dst[i] = *(const unsigned int*)(ap + i * 2);
            } else {
                #pragma unroll
                for (int i = 0; i < 8; ++i) dst[i] = 0u;
            }
        }
        {   // B tile (same shape, [N][K])
            const int r  = tid >> 1;
            const int c0 = (tid & 1) * 16;
            unsigned int* dst = (unsigned int*)&Bs[cur][r][c0];
            if (nBase + r < N) {
                const unsigned short* bp = bTile + (long)r * ldb + k0 + c0;
                #pragma unroll
                for (int i = 0; i < 8; ++i) dst[i] = *(const unsigned int*)(bp + i * 2);
            } else {
                #pragma unroll
                for (int i = 0; i < 8; ++i) dst[i] = 0u;
            }
        }
        __syncthreads();
        wmma_step(&As[cur][0][0], &Bs[cur][0][0], wm, wn, lm, lh, acc);
        __syncthreads();
    }
#endif

    // ---- epilogue ----
    #pragma unroll
    for (int im = 0; im < 2; ++im) {
        #pragma unroll
        for (int in_ = 0; in_ < 4; ++in_) {
            #pragma unroll
            for (int v = 0; v < 8; ++v) {
                const int row = mBase + wm * 32 + im * 16 + lh * 8 + v;
                const int col = nBase + wn * 64 + in_ * 16 + lm;
                if (row < M && col < N) {
                    float val = acc[im][in_][v] * alpha;
                    if (CT) {
                        const long cidx = offC + (long)col * ldc + row;
                        ((unsigned short*)Cout)[cidx] = f2bf(val);
                    } else {
                        const long cidx = offC + (long)row * ldc + col;
                        if (Gate) {
                            const long grow = (offC / ldc + row) / gdiv;
                            val *= Gate[grow * (long)gstride + col];
                        }
                        if (Res) val += Res[cidx];
                        if (OUT_BF16) ((unsigned short*)Cout)[cidx] = f2bf(val);
                        else          ((float*)Cout)[cidx] = val;
                    }
                }
            }
        }
    }
}

// ---------------------------------------------------------------------------
// fp32 -> bf16 TRANSPOSING weight conversion: out[n*K + k] = bf16(in[k*N + n])
// ---------------------------------------------------------------------------
__global__ __launch_bounds__(256)
void cvt_t_bf16(const float* __restrict__ in, unsigned short* __restrict__ out,
                int K, int N) {
    __shared__ float t[32][33];
    const int kb = blockIdx.y * 32, nb = blockIdx.x * 32;
    const int tx = threadIdx.x & 31, ty = threadIdx.x >> 5;
    for (int i = ty; i < 32; i += 8) {
        const int k = kb + i, n = nb + tx;
        t[i][tx] = (k < K && n < N) ? in[(long)k * N + n] : 0.0f;
    }
    __syncthreads();
    for (int i = ty; i < 32; i += 8) {
        const int n = nb + i, k = kb + tx;
        if (n < N && k < K) out[(long)n * K + k] = f2bf(t[tx][i]);
    }
}

// y = rms(x) * (1 + scale[(row/sdiv)*sstride + i]) -> bf16
__global__ __launch_bounds__(256)
void rms_bf16(const float* __restrict__ x, unsigned short* __restrict__ y,
              const float* __restrict__ scale, int D, int sdiv, int sstride) {
    const int row = blockIdx.x;
    const int tid = threadIdx.x;
    const long base = (long)row * D;
    float ss = 0.0f;
    for (int i = tid; i < D; i += 256) { float v = x[base + i]; ss += v * v; }
    __shared__ float red[256];
    red[tid] = ss; __syncthreads();
    for (int o = 128; o > 0; o >>= 1) { if (tid < o) red[tid] += red[tid + o]; __syncthreads(); }
    const float inv = rsqrtf(red[0] / (float)D + 1e-6f);
    const long soff = (long)(row / sdiv) * sstride;
    for (int i = tid; i < D; i += 256)
        y[base + i] = f2bf(x[base + i] * inv * (1.0f + scale[soff + i]));
}

// out[b][n] = bias[n] + sum_k cond[b][k] * W[k][n]   (tiny M=4 projection)
__global__ void ada_proj(const float* __restrict__ cond, const float* __restrict__ W,
                         const float* __restrict__ bias, float* __restrict__ out,
                         int Bn, int K, int N) {
    int idx = blockIdx.x * blockDim.x + threadIdx.x;
    if (idx >= Bn * N) return;
    const int b = idx / N, n = idx % N;
    float acc = bias[n];
    for (int k = 0; k < K; ++k) acc += cond[(long)b * K + k] * W[(long)k * N + n];
    out[idx] = acc;
}

__global__ void cumsum_masks(const int* __restrict__ pad, const int* __restrict__ att,
                             int* __restrict__ pos, int* __restrict__ cs, int Bn, int S) {
    const int b = threadIdx.x;
    if (b >= Bn) return;
    int p = 0, c = 0;
    for (int s = 0; s < S; ++s) {
        p += pad[b * S + s]; c += att[b * S + s];
        pos[b * S + s] = p - 1; cs[b * S + s] = c;
    }
}

// RoPE: in fp32 (B,S,Hn,Hd) -> out bf16, half-rotation pairs (j, j+Hd/2)
__global__ void rope_bf16(const float* __restrict__ in, unsigned short* __restrict__ out,
                          const int* __restrict__ pos, int Bn, int S, int Hn, int Hd) {
    const int h2 = Hd / 2;
    long idx = (long)blockIdx.x * blockDim.x + threadIdx.x;
    const long total = (long)Bn * S * Hn * h2;
    if (idx >= total) return;
    const int j = (int)(idx % h2); long t = idx / h2;
    const int h = (int)(t % Hn); t /= Hn;
    const int s = (int)(t % S);  const int b = (int)(t / S);
    const long base = (((long)b * S + s) * Hn + h) * Hd;
    const float p  = (float)pos[b * S + s];
    const float fr = p * __powf(10000.0f, -2.0f * (float)j / (float)Hd);
    const float c = __cosf(fr), sn = __sinf(fr);
    const float a0 = in[base + j], a1 = in[base + j + h2];
    out[base + j]      = f2bf(a0 * c - a1 * sn);
    out[base + j + h2] = f2bf(a1 * c + a0 * sn);
}

// Masked softmax over a row of length S; scores fp32 -> probs bf16
__global__ __launch_bounds__(256)
void softmax_bf16(const float* __restrict__ sc, unsigned short* __restrict__ pr,
                  const int* __restrict__ cs, const int* __restrict__ pad,
                  int Hn, int S) {
    const int rowId = blockIdx.x;              // (b*Hn + h)*S + q
    const int tid   = threadIdx.x;
    const int q  = rowId % S;
    const int bh = rowId / S;
    const int b  = bh / Hn;
    const long base = (long)rowId * S;
    const int csq = cs[b * S + q];
    const int pq  = pad[b * S + q];
    __shared__ float red[256];

    float mx = -3.4e38f;
    for (int k = tid; k < S; k += 256) {
        const int ok = (cs[b * S + k] <= csq) && pq && pad[b * S + k];
        const float v = ok ? sc[base + k] : MASK_VAL;
        mx = fmaxf(mx, v);
    }
    red[tid] = mx; __syncthreads();
    for (int o = 128; o > 0; o >>= 1) { if (tid < o) red[tid] = fmaxf(red[tid], red[tid + o]); __syncthreads(); }
    const float rowmax = red[0]; __syncthreads();

    float sum = 0.0f;
    for (int k = tid; k < S; k += 256) {
        const int ok = (cs[b * S + k] <= csq) && pq && pad[b * S + k];
        const float v = ok ? sc[base + k] : MASK_VAL;
        sum += __expf(v - rowmax);
    }
    red[tid] = sum; __syncthreads();
    for (int o = 128; o > 0; o >>= 1) { if (tid < o) red[tid] += red[tid + o]; __syncthreads(); }
    const float inv = 1.0f / red[0];

    for (int k = tid; k < S; k += 256) {
        const int ok = (cs[b * S + k] <= csq) && pq && pad[b * S + k];
        const float v = ok ? sc[base + k] : MASK_VAL;
        pr[base + k] = f2bf(__expf(v - rowmax) * inv);
    }
}

// t = gelu_tanh(g) * u   (bf16 in, bf16 out; may alias out==g)
__global__ void gelu_mul_bf16(const unsigned short* __restrict__ g,
                              const unsigned short* __restrict__ u,
                              unsigned short* __restrict__ out, long n) {
    long i = (long)blockIdx.x * blockDim.x + threadIdx.x;
    if (i >= n) return;
    const float x  = bf2f(g[i]);
    const float uu = bf2f(u[i]);
    const float t  = 0.7978845608028654f * (x + 0.044715f * x * x * x);
    out[i] = f2bf(0.5f * x * (1.0f + tanhf(t)) * uu);
}

// ---------------------------------------------------------------------------
// Host-side launch helpers
// ---------------------------------------------------------------------------
template<bool OB, bool CT>
static void gemm(hipStream_t st, const unsigned short* A, const unsigned short* B, void* C,
                 const float* Res, const float* Gate,
                 int M, int N, int K, int lda, int ldb, int ldc, float alpha,
                 int Z, int divA, long sAb, long sAh,
                 int divB, long sBb, long sBh,
                 int divC, long sCb, long sCh,
                 int gdiv, int gstride) {
    dim3 grid((N + 127) / 128, (M + 127) / 128, Z), blk(256);
    gemm_wmma<OB, CT><<<grid, blk, 0, st>>>(A, B, C, Res, Gate, M, N, K, lda, ldb, ldc, alpha,
                                            divA, sAb, sAh, divB, sBb, sBh, divC, sCb, sCh,
                                            gdiv, gstride);
}

static void cvtT(hipStream_t st, const float* in, unsigned short* out, int K, int N) {
    dim3 grid((N + 31) / 32, (K + 31) / 32), blk(256);
    cvt_t_bf16<<<grid, blk, 0, st>>>(in, out, K, N);
}

extern "C" void kernel_launch(void* const* d_in, const int* in_sizes, int n_in,
                              void* d_out, int out_size, void* d_ws, size_t ws_size,
                              hipStream_t stream) {
    (void)in_sizes; (void)n_in; (void)out_size; (void)ws_size;
    // ---- inputs (setup_inputs order) ----
    const float* x0        = (const float*)d_in[0];
    const float* x1        = (const float*)d_in[1];
    const float* cond1     = (const float*)d_in[2];
    const float* w_q0      = (const float*)d_in[3];
    const float* w_k0      = (const float*)d_in[4];
    const float* w_v0      = (const float*)d_in[5];
    const float* w_o0      = (const float*)d_in[6];
    const float* norm1_w0  = (const float*)d_in[7];
    const float* norm2_w0  = (const float*)d_in[8];
    const float* w_gate0   = (const float*)d_in[9];
    const float* w_up0     = (const float*)d_in[10];
    const float* w_down0   = (const float*)d_in[11];
    const float* w_q1      = (const float*)d_in[12];
    const float* w_k1      = (const float*)d_in[13];
    const float* w_v1      = (const float*)d_in[14];
    const float* w_o1      = (const float*)d_in[15];
    const float* ada_in_w1 = (const float*)d_in[16];
    const float* ada_in_b1 = (const float*)d_in[17];
    const float* ada_post_w1 = (const float*)d_in[18];
    const float* ada_post_b1 = (const float*)d_in[19];
    const float* w_gate1   = (const float*)d_in[20];
    const float* w_up1     = (const float*)d_in[21];
    const float* w_down1   = (const float*)d_in[22];
    const int*   pad_masks = (const int*)d_in[23];
    const int*   att_masks = (const int*)d_in[24];

    float* out0 = (float*)d_out;                       // (B,S1,D0)
    float* out1 = out0 + (long)BB * S1N * D0N;         // (B,S2,D1)

    // ---- workspace allocator ----
    char* wsp = (char*)d_ws;
    size_t off = 0;
    auto alloc = [&](size_t bytes) -> char* {
        char* p = wsp + off;
        off += (bytes + 255) & ~(size_t)255;
        return p;
    };
    auto allocB = [&](long n) { return (unsigned short*)alloc((size_t)n * 2); };
    auto allocF = [&](long n) { return (float*)alloc((size_t)n * 4); };
    auto allocI = [&](long n) { return (int*)alloc((size_t)n * 4); };

    // bf16 weight copies, ALL transposed to [N][K] (k-contiguous for TDM tiles)
    unsigned short* wq0b = allocB((long)D0N * HN * HDN);   // [2048][D0]
    unsigned short* wk0b = allocB((long)D0N * HDN);        // [256][D0]
    unsigned short* wv0b = allocB((long)D0N * HDN);
    unsigned short* wo0b = allocB((long)HN * HDN * D0N);   // [D0][2048]
    unsigned short* wg0b = allocB((long)D0N * F0N);        // [F0][D0]
    unsigned short* wu0b = allocB((long)D0N * F0N);
    unsigned short* wd0b = allocB((long)F0N * D0N);        // [D0][F0]
    unsigned short* wq1b = allocB((long)D1N * HN * HDN);
    unsigned short* wk1b = allocB((long)D1N * HDN);
    unsigned short* wv1b = allocB((long)D1N * HDN);
    unsigned short* wo1b = allocB((long)HN * HDN * D1N);
    unsigned short* wg1b = allocB((long)D1N * F1N);
    unsigned short* wu1b = allocB((long)D1N * F1N);
    unsigned short* wd1b = allocB((long)F1N * D1N);
    // activations
    unsigned short* h0b  = allocB((long)BB * S1N * D0N);
    unsigned short* h1b  = allocB((long)BB * S2N * D1N);
    float* mod_in   = allocF((long)BB * 2 * D1N);
    float* mod_post = allocF((long)BB * 2 * D1N);
    int*   posb = allocI((long)BB * SN);
    int*   csb  = allocI((long)BB * SN);
    float* qf = allocF((long)BB * SN * HN * HDN);
    float* kf = allocF((long)BB * SN * HDN);
    unsigned short* vbT = allocB((long)BB * HDN * SN);     // transposed: (B, HD, S)
    unsigned short* qb  = allocB((long)BB * SN * HN * HDN);
    unsigned short* kb  = allocB((long)BB * SN * HDN);
    float* scoresf = allocF((long)BB * HN * SN * SN);
    unsigned short* probsb = allocB((long)BB * HN * SN * SN);
    unsigned short* attb = allocB((long)BB * SN * HN * HDN);
    float* r0f = allocF((long)BB * S1N * D0N);
    float* r1f = allocF((long)BB * S2N * D1N);
    unsigned short* y0b = allocB((long)BB * S1N * D0N);
    unsigned short* y1b = allocB((long)BB * S2N * D1N);
    unsigned short* g0b = allocB((long)BB * S1N * F0N);
    unsigned short* u0b = allocB((long)BB * S1N * F0N);
    unsigned short* g1b = allocB((long)BB * S2N * F1N);
    unsigned short* u1b = allocB((long)BB * S2N * F1N);

    // ---- 1) weight conversion to bf16 + transpose ----
    cvtT(stream, w_q0,   wq0b, D0N, HN*HDN);
    cvtT(stream, w_k0,   wk0b, D0N, HDN);
    cvtT(stream, w_v0,   wv0b, D0N, HDN);
    cvtT(stream, w_o0,   wo0b, HN*HDN, D0N);
    cvtT(stream, w_gate0,wg0b, D0N, F0N);
    cvtT(stream, w_up0,  wu0b, D0N, F0N);
    cvtT(stream, w_down0,wd0b, F0N, D0N);
    cvtT(stream, w_q1,   wq1b, D1N, HN*HDN);
    cvtT(stream, w_k1,   wk1b, D1N, HDN);
    cvtT(stream, w_v1,   wv1b, D1N, HDN);
    cvtT(stream, w_o1,   wo1b, HN*HDN, D1N);
    cvtT(stream, w_gate1,wg1b, D1N, F1N);
    cvtT(stream, w_up1,  wu1b, D1N, F1N);
    cvtT(stream, w_down1,wd1b, F1N, D1N);

    // ---- 2) masks / adaLN ----
    cumsum_masks<<<1, BB, 0, stream>>>(pad_masks, att_masks, posb, csb, BB, SN);
    ada_proj<<<(BB * 2 * D1N + 255) / 256, 256, 0, stream>>>(cond1, ada_in_w1,   ada_in_b1,   mod_in,   BB, D1N, 2 * D1N);
    ada_proj<<<(BB * 2 * D1N + 255) / 256, 256, 0, stream>>>(cond1, ada_post_w1, ada_post_b1, mod_post, BB, D1N, 2 * D1N);

    // ---- 3) pre-norms ----
    rms_bf16<<<BB * S1N, 256, 0, stream>>>(x0, h0b, norm1_w0, D0N, 1, 0);
    rms_bf16<<<BB * S2N, 256, 0, stream>>>(x1, h1b, mod_in,   D1N, S2N, 2 * D1N);  // scale_in

    // ---- 4) QKV projections (batched over z=b) ----
    const long qRow = (long)HN * HDN;        // 2048
    gemm<false,false>(stream, h0b, wq0b, qf, nullptr, nullptr,
        S1N, HN*HDN, D0N, D0N, D0N, HN*HDN, 1.0f, BB,
        1, (long)S1N*D0N, 0,  1, 0, 0,  1, (long)SN*qRow, 0,  1, 0);
    gemm<false,false>(stream, h1b, wq1b, qf + (long)S1N*qRow, nullptr, nullptr,
        S2N, HN*HDN, D1N, D1N, D1N, HN*HDN, 1.0f, BB,
        1, (long)S2N*D1N, 0,  1, 0, 0,  1, (long)SN*qRow, 0,  1, 0);
    gemm<false,false>(stream, h0b, wk0b, kf, nullptr, nullptr,
        S1N, HDN, D0N, D0N, D0N, HDN, 1.0f, BB,
        1, (long)S1N*D0N, 0,  1, 0, 0,  1, (long)SN*HDN, 0,  1, 0);
    gemm<false,false>(stream, h1b, wk1b, kf + (long)S1N*HDN, nullptr, nullptr,
        S2N, HDN, D1N, D1N, D1N, HDN, 1.0f, BB,
        1, (long)S2N*D1N, 0,  1, 0, 0,  1, (long)SN*HDN, 0,  1, 0);
    // v -> TRANSPOSED bf16 store (B, HD, S): feeds att GEMM's [N][K] B operand
    gemm<true,true>(stream, h0b, wv0b, vbT, nullptr, nullptr,
        S1N, HDN, D0N, D0N, D0N, SN, 1.0f, BB,
        1, (long)S1N*D0N, 0,  1, 0, 0,  1, (long)HDN*SN, 0,  1, 0);
    gemm<true,true>(stream, h1b, wv1b, vbT + S1N, nullptr, nullptr,
        S2N, HDN, D1N, D1N, D1N, SN, 1.0f, BB,
        1, (long)S2N*D1N, 0,  1, 0, 0,  1, (long)HDN*SN, 0,  1, 0);

    // ---- 5) RoPE (fp32 -> bf16) ----
    {
        long tq = (long)BB * SN * HN * (HDN / 2);
        rope_bf16<<<(unsigned)((tq + 255) / 256), 256, 0, stream>>>(qf, qb, posb, BB, SN, HN, HDN);
        long tk = (long)BB * SN * 1 * (HDN / 2);
        rope_bf16<<<(unsigned)((tk + 255) / 256), 256, 0, stream>>>(kf, kb, posb, BB, SN, 1, HDN);
    }

    // ---- 6) scores = (q @ k^T) * HD^-0.5, z = b*H + h ----
    // B operand = kb (B,S,HD): element [k=d][n=s_k] at s_k*HD + d -> [N][K] layout
    gemm<false,false>(stream, qb, kb, scoresf, nullptr, nullptr,
        SN, SN, HDN, HN*HDN, HDN, SN, 0.0625f, BB*HN,
        HN, (long)SN*qRow, (long)HDN,
        HN, (long)SN*HDN, 0,
        1,  (long)SN*SN, 0,  1, 0);

    // ---- 7) masked softmax -> bf16 probs ----
    softmax_bf16<<<BB * HN * SN, 256, 0, stream>>>(scoresf, probsb, csb, pad_masks, HN, SN);

    // ---- 8) att = probs @ v -> attb (B,S,H*HD) bf16 ----
    // B operand = vbT (B,HD,S): element [k=s_k][n=d] at d*SN + s_k -> [N][K] layout
    gemm<true,false>(stream, probsb, vbT, attb, nullptr, nullptr,
        SN, HDN, SN, SN, SN, HN*HDN, 1.0f, BB*HN,
        1,  (long)SN*SN, 0,
        HN, (long)HDN*SN, 0,
        HN, (long)SN*qRow, (long)HDN,  1, 0);

    // ---- 9) output projections + residual (+gate for stream 1) ----
    gemm<false,false>(stream, attb, wo0b, r0f, x0, nullptr,
        S1N, D0N, HN*HDN, HN*HDN, HN*HDN, D0N, 1.0f, BB,
        1, (long)SN*qRow, 0,  1, 0, 0,  1, (long)S1N*D0N, 0,  1, 0);
    gemm<false,false>(stream, attb + (long)S1N*qRow, wo1b, r1f, x1, mod_in + D1N,
        S2N, D1N, HN*HDN, HN*HDN, HN*HDN, D1N, 1.0f, BB,
        1, (long)SN*qRow, 0,  1, 0, 0,  1, (long)S2N*D1N, 0,  S2N, 2*D1N);

    // ---- 10) post norms ----
    rms_bf16<<<BB * S1N, 256, 0, stream>>>(r0f, y0b, norm2_w0, D0N, 1, 0);
    rms_bf16<<<BB * S2N, 256, 0, stream>>>(r1f, y1b, mod_post, D1N, S2N, 2 * D1N);  // scale_post

    // ---- 11) MLP0 ----
    gemm<true,false>(stream, y0b, wg0b, g0b, nullptr, nullptr,
        BB*S1N, F0N, D0N, D0N, D0N, F0N, 1.0f, 1, 1,0,0, 1,0,0, 1,0,0, 1,0);
    gemm<true,false>(stream, y0b, wu0b, u0b, nullptr, nullptr,
        BB*S1N, F0N, D0N, D0N, D0N, F0N, 1.0f, 1, 1,0,0, 1,0,0, 1,0,0, 1,0);
    {
        long n = (long)BB * S1N * F0N;
        gelu_mul_bf16<<<(unsigned)((n + 255) / 256), 256, 0, stream>>>(g0b, u0b, g0b, n);
    }
    gemm<false,false>(stream, g0b, wd0b, out0, r0f, nullptr,
        BB*S1N, D0N, F0N, F0N, F0N, D0N, 1.0f, 1, 1,0,0, 1,0,0, 1,0,0, 1,0);

    // ---- 12) MLP1 ----
    gemm<true,false>(stream, y1b, wg1b, g1b, nullptr, nullptr,
        BB*S2N, F1N, D1N, D1N, D1N, F1N, 1.0f, 1, 1,0,0, 1,0,0, 1,0,0, 1,0);
    gemm<true,false>(stream, y1b, wu1b, u1b, nullptr, nullptr,
        BB*S2N, F1N, D1N, D1N, D1N, F1N, 1.0f, 1, 1,0,0, 1,0,0, 1,0,0, 1,0);
    {
        long n = (long)BB * S2N * F1N;
        gelu_mul_bf16<<<(unsigned)((n + 255) / 256), 256, 0, stream>>>(g1b, u1b, g1b, n);
    }
    gemm<false,false>(stream, g1b, wd1b, out1, r1f, mod_post + D1N,
        BB*S2N, D1N, F1N, F1N, F1N, D1N, 1.0f, 1,
        1,0,0, 1,0,0, 1,0,0, S2N, 2*D1N);
}
